// GPoolBlock_42374147342988
// MI455X (gfx1250) — matrix operand order repaired and checked by reading, not compiled, not run on hardware
//
#include <hip/hip_runtime.h>

#define BATCH   256
#define HW      361
#define TRUNK_C 256
#define REG_C   192
#define GPOOL_C 64

typedef __attribute__((ext_vector_type(16))) __bf16 v16bf;
typedef __attribute__((ext_vector_type(8)))  float  v8f;

__device__ __forceinline__ unsigned short f2bf(float f) {
    unsigned int u = __float_as_uint(f);
    unsigned int r = u + 0x7FFFu + ((u >> 16) & 1u);   // round-to-nearest-even
    return (unsigned short)(r >> 16);
}

__device__ __forceinline__ float mishf(float x) {
    float sp = (x > 20.f) ? x : log1pf(__expf(x));
    return x * tanhf(sp);
}

// Async copy 16B from global (saddr + 32-bit voffset) straight into LDS.
// vdst = per-lane LDS byte offset (low 32 bits of flat shared pointer).
// Tracked by ASYNCcnt; drained with s_wait_asynccnt.
#define ASYNC_LD_B128(ldsOff, vOff, sBase)                                  \
    asm volatile("global_load_async_to_lds_b128 %0, %1, %2"                 \
                 :: "v"(ldsOff), "v"(vOff), "s"(sBase) : "memory")
#define ASYNC_WAIT() asm volatile("s_wait_asynccnt 0x0" ::: "memory")

// ---------------------------------------------------------------------------
// Stage 1: act1[b][p][c] = bf16(mish(bn1(x[b][c][p])))      (NHWC, bf16)
// ---------------------------------------------------------------------------
__global__ __launch_bounds__(256) void k_bn_mish_nhwc(
    const float* __restrict__ x, const float* __restrict__ g,
    const float* __restrict__ bt, const float* __restrict__ mn,
    const float* __restrict__ vr, unsigned short* __restrict__ out,
    long long total) {
    long long i = (long long)blockIdx.x * blockDim.x + threadIdx.x;
    if (i >= total) return;
    int c = (int)(i % TRUNK_C);
    long long rest = i / TRUNK_C;
    int p = (int)(rest % HW);
    long long b = rest / HW;
    float inv = g[c] * rsqrtf(vr[c] + 1e-5f);
    float val = (x[(b * TRUNK_C + c) * HW + p] - mn[c]) * inv + bt[c];
    out[i] = f2bf(mishf(val));
}

// ---------------------------------------------------------------------------
// Weight reorder + convert: OIHW f32 -> [rowOff+oc][kidx][ic] bf16
// ---------------------------------------------------------------------------
__global__ __launch_bounds__(256) void k_wreorder(
    const float* __restrict__ in, unsigned short* __restrict__ out,
    int OC, int IC, int rowOff) {
    int i = blockIdx.x * 256 + threadIdx.x;
    if (i >= OC * IC * 9) return;
    int oc = i / (IC * 9);
    int r  = i - oc * (IC * 9);
    int kidx = r / IC;
    int ic   = r - kidx * IC;
    out[((size_t)(rowOff + oc) * 9 + kidx) * IC + ic] =
        f2bf(in[((size_t)oc * IC + ic) * 9 + kidx]);
}

// ---------------------------------------------------------------------------
// 3x3 SAME conv on 19x19 as implicit GEMM, bf16 WMMA, f32 accumulate.
//   A = weights [OC][9][Cin]; B = im2col rows from NHWC act (contiguous 128B).
// Block: 256 thr (8 waves). Tile M=128 x N=32, K-step 64.
// Double-buffered LDS; async-to-LDS staging software-pipelined one chunk
// ahead (issue after barrier, drain at next iteration top). 4 WMMA / barrier.
// Output rows < splitC -> outA, rest -> outB (fused dual-head conv).
// ---------------------------------------------------------------------------
#define MT 128
#define NT 32
#define KC 64

__global__ __launch_bounds__(256) void k_conv3x3_wmma(
    const unsigned short* __restrict__ act,   // bf16 NHWC [B][361][Cin]
    const unsigned short* __restrict__ wgt,   // bf16 [OC][9][Cin]
    float* __restrict__ outA,                 // f32 NCHW [B][splitC][361]
    float* __restrict__ outB,                 // f32 NCHW [B][OC-splitC][361]
    const float* __restrict__ residual,       // optional (same layout as outA)
    int Cin, int OC, int splitC) {
    const int K   = Cin * 9;
    const int b   = blockIdx.z;
    const int oc0 = blockIdx.y * MT;
    const int n0  = blockIdx.x * NT;
    const int tid = threadIdx.x;

    __shared__ __align__(16) unsigned short As[2][MT][KC];   // 32 KB
    __shared__ __align__(16) unsigned short Bs[2][NT][KC];   //  8 KB

    const int wave = tid >> 5;        // 0..7 -> M offset 16*wave
    const int lane = tid & 31;
    const int half = lane >> 4;
    const int l15  = lane & 15;

    v8f acc0 = {};
    v8f acc1 = {};

    const unsigned short* actB = act + (size_t)b * HW * Cin;

    // A staging: 128x64 bf16 = 16KB; thread owns 64B (4 x b128) of one row
    const int a_row = tid >> 1;            // 0..127
    const int a_col = (tid & 1) * 32;      // 0 or 32 (elements)
    // B staging: 32 rows x 128B; thread owns 16B (1 x b128)
    const int b_n = tid >> 3;              // 0..31 (spatial within tile)
    const int b_c = (tid & 7) * 8;         // channel sub-offset

    const unsigned ldsAbase = (unsigned)(size_t)&As[0][0][0] +
                              (unsigned)(a_row * KC + a_col) * 2u;
    const unsigned ldsBbase = (unsigned)(size_t)&Bs[0][0][0] +
                              (unsigned)(b_n * KC + b_c) * 2u;

    const int p  = n0 + b_n;
    const int py = p / 19;
    const int px = p - py * 19;

    auto issue = [&](int kidx_, int ic0_, int s_) {
        const int ky = kidx_ / 3;
        const int kx = kidx_ - 3 * ky;
        const int iy = py + ky - 1;
        const int ix = px + kx - 1;
        const bool valid = (p < HW) && ((unsigned)iy < 19u) && ((unsigned)ix < 19u);

        unsigned aoff = ((unsigned)(oc0 + a_row) * (unsigned)K +
                         (unsigned)kidx_ * (unsigned)Cin +
                         (unsigned)(ic0_ + a_col)) * 2u;
        unsigned ldsA = ldsAbase + (unsigned)s_ * (MT * KC * 2u);
        ASYNC_LD_B128(ldsA,       aoff,       wgt);
        ASYNC_LD_B128(ldsA + 16u, aoff + 16u, wgt);
        ASYNC_LD_B128(ldsA + 32u, aoff + 32u, wgt);
        ASYNC_LD_B128(ldsA + 48u, aoff + 48u, wgt);

        if (valid) {
            unsigned boff = ((unsigned)(iy * 19 + ix) * (unsigned)Cin +
                             (unsigned)(ic0_ + b_c)) * 2u;
            ASYNC_LD_B128(ldsBbase + (unsigned)s_ * (NT * KC * 2u), boff, actB);
        } else {
            *(uint4*)&Bs[s_][b_n][b_c] = make_uint4(0u, 0u, 0u, 0u);
        }
    };

    // ---- software pipeline: prologue stages chunk 0 into buffer 0 ----
    issue(0, 0, 0);
    int kidx = 0, ic0 = 0, sel = 0;

    for (;;) {
        int kidxN = kidx, ic0N = ic0 + KC;
        if (ic0N == Cin) { ic0N = 0; kidxN = kidx + 1; }

        ASYNC_WAIT();            // drain current chunk (next not yet issued)
        __syncthreads();         // all waves' data landed; prev reads done
        if (kidxN < 9) issue(kidxN, ic0N, sel ^ 1);   // run ahead one chunk

        // ---- fragments (contiguous k = 16*half + s, same for A and B) ----
        const unsigned short (*A)[KC] = As[sel];
        const unsigned short (*Bt)[KC] = Bs[sel];
        union { v16bf v; uint4 q[2]; } a0, a1, b0, b1, b2, b3;
        const int ar = wave * 16 + l15;
        a0.q[0] = *(const uint4*)&A[ar][half * 16];
        a0.q[1] = *(const uint4*)&A[ar][half * 16 + 8];
        a1.q[0] = *(const uint4*)&A[ar][32 + half * 16];
        a1.q[1] = *(const uint4*)&A[ar][32 + half * 16 + 8];
        b0.q[0] = *(const uint4*)&Bt[l15][half * 16];
        b0.q[1] = *(const uint4*)&Bt[l15][half * 16 + 8];
        b1.q[0] = *(const uint4*)&Bt[l15][32 + half * 16];
        b1.q[1] = *(const uint4*)&Bt[l15][32 + half * 16 + 8];
        b2.q[0] = *(const uint4*)&Bt[16 + l15][half * 16];
        b2.q[1] = *(const uint4*)&Bt[16 + l15][half * 16 + 8];
        b3.q[0] = *(const uint4*)&Bt[16 + l15][32 + half * 16];
        b3.q[1] = *(const uint4*)&Bt[16 + l15][32 + half * 16 + 8];

        acc0 = __builtin_amdgcn_wmma_f32_16x16x32_bf16(
            false, a0.v, false, b0.v, (short)0, acc0, false, false);
        acc0 = __builtin_amdgcn_wmma_f32_16x16x32_bf16(
            false, a1.v, false, b1.v, (short)0, acc0, false, false);
        acc1 = __builtin_amdgcn_wmma_f32_16x16x32_bf16(
            false, a0.v, false, b2.v, (short)0, acc1, false, false);
        acc1 = __builtin_amdgcn_wmma_f32_16x16x32_bf16(
            false, a1.v, false, b3.v, (short)0, acc1, false, false);

        if (kidxN >= 9) break;
        kidx = kidxN; ic0 = ic0N; sel ^= 1;
    }

    // ---- epilogue: C/D layout: vgpr r -> M = r + 8*(lane>=16), N = lane&15
    #pragma unroll
    for (int t = 0; t < 2; ++t) {
        const v8f& acc = t ? acc1 : acc0;
        const int n_g = n0 + t * 16 + l15;
        if (n_g < HW) {
            #pragma unroll
            for (int r = 0; r < 8; ++r) {
                int ocg = oc0 + wave * 16 + half * 8 + r;
                float v = acc[r];
                if (ocg < splitC) {
                    size_t idx = ((size_t)b * splitC + ocg) * HW + n_g;
                    if (residual) v += residual[idx];
                    outA[idx] = v;
                } else {
                    size_t idx = ((size_t)b * (OC - splitC) + (ocg - splitC)) * HW + n_g;
                    outB[idx] = v;
                }
            }
        }
    }
}

// ---------------------------------------------------------------------------
// gpool path: bn1b + mish on the fly, then {mean, mean*0.5, max} -> [B, 192]
// ---------------------------------------------------------------------------
__global__ __launch_bounds__(128) void k_gp_pool(
    const float* __restrict__ gpc, const float* __restrict__ g,
    const float* __restrict__ bt, const float* __restrict__ mn,
    const float* __restrict__ vr, float* __restrict__ gpool) {
    int bc = blockIdx.x;          // b*64 + c
    int c  = bc & 63;
    int b  = bc >> 6;
    float inv = g[c] * rsqrtf(vr[c] + 1e-5f);
    float sum = 0.f, mx = -1e30f;
    const float* src = gpc + (size_t)bc * HW;
    for (int i = threadIdx.x; i < HW; i += 128) {
        float v = mishf((src[i] - mn[c]) * inv + bt[c]);
        sum += v;
        mx = fmaxf(mx, v);
    }
    __shared__ float ss[128], sm[128];
    ss[threadIdx.x] = sum;
    sm[threadIdx.x] = mx;
    __syncthreads();
    for (int s = 64; s > 0; s >>= 1) {
        if (threadIdx.x < s) {
            ss[threadIdx.x] += ss[threadIdx.x + s];
            sm[threadIdx.x] = fmaxf(sm[threadIdx.x], sm[threadIdx.x + s]);
        }
        __syncthreads();
    }
    if (threadIdx.x == 0) {
        float mean = ss[0] * (1.f / 361.f);
        gpool[(size_t)b * (3 * GPOOL_C) + c]               = mean;
        gpool[(size_t)b * (3 * GPOOL_C) + GPOOL_C + c]     = mean * 0.5f;
        gpool[(size_t)b * (3 * GPOOL_C) + 2 * GPOOL_C + c] = sm[0];
    }
}

// ---------------------------------------------------------------------------
// gpbias[b][oc] = gpool[b][:] . w_lin[oc][:]
// ---------------------------------------------------------------------------
__global__ __launch_bounds__(256) void k_gp_linear(
    const float* __restrict__ gpool, const float* __restrict__ wlin,
    float* __restrict__ gpbias) {
    int i = blockIdx.x * 256 + threadIdx.x;     // b*192 + oc
    if (i >= BATCH * REG_C) return;
    int oc = i % REG_C;
    int b  = i / REG_C;
    const float* gv = gpool + (size_t)b * (3 * GPOOL_C);
    const float* wv = wlin + (size_t)oc * (3 * GPOOL_C);
    float s = 0.f;
    #pragma unroll 4
    for (int j = 0; j < 3 * GPOOL_C; ++j) s += gv[j] * wv[j];
    gpbias[i] = s;
}

// ---------------------------------------------------------------------------
// act2[b][p][c] = bf16(mish(bn2(reg[b][c][p] + gpbias[b][c])))   (NHWC)
// ---------------------------------------------------------------------------
__global__ __launch_bounds__(256) void k_bias_bn_mish_nhwc(
    const float* __restrict__ reg, const float* __restrict__ gpbias,
    const float* __restrict__ g, const float* __restrict__ bt,
    const float* __restrict__ mn, const float* __restrict__ vr,
    unsigned short* __restrict__ out, long long total) {
    long long i = (long long)blockIdx.x * blockDim.x + threadIdx.x;
    if (i >= total) return;
    int c = (int)(i % REG_C);
    long long rest = i / REG_C;
    int p = (int)(rest % HW);
    long long b = rest / HW;
    long long bo = b * REG_C + c;
    float inv = g[c] * rsqrtf(vr[c] + 1e-5f);
    float val = reg[bo * HW + p] + gpbias[bo];
    out[i] = f2bf(mishf((val - mn[c]) * inv + bt[c]));
}

// ---------------------------------------------------------------------------
extern "C" void kernel_launch(void* const* d_in, const int* in_sizes, int n_in,
                              void* d_out, int out_size, void* d_ws, size_t ws_size,
                              hipStream_t stream) {
    const float* x     = (const float*)d_in[0];
    const float* bn1_g = (const float*)d_in[1];
    const float* bn1_b = (const float*)d_in[2];
    const float* bn1_m = (const float*)d_in[3];
    const float* bn1_v = (const float*)d_in[4];
    const float* w1a   = (const float*)d_in[5];
    const float* w1b   = (const float*)d_in[6];
    const float* b1b_g = (const float*)d_in[7];
    const float* b1b_b = (const float*)d_in[8];
    const float* b1b_m = (const float*)d_in[9];
    const float* b1b_v = (const float*)d_in[10];
    const float* w_lin = (const float*)d_in[11];
    const float* bn2_g = (const float*)d_in[12];
    const float* bn2_b = (const float*)d_in[13];
    const float* bn2_m = (const float*)d_in[14];
    const float* bn2_v = (const float*)d_in[15];
    const float* w2    = (const float*)d_in[16];

    char* ws = (char*)d_ws;
    size_t off = 0;
    auto alloc = [&](size_t bytes) -> void* {
        void* pp = ws + off;
        off = (off + bytes + 255) & ~(size_t)255;
        return pp;
    };

    const long long actN = (long long)BATCH * TRUNK_C * HW;
    const long long regN = (long long)BATCH * REG_C * HW;
    const long long gpcN = (long long)BATCH * GPOOL_C * HW;

    unsigned short* act1   = (unsigned short*)alloc(actN * 2);                 // NHWC bf16
    unsigned short* wfused = (unsigned short*)alloc((size_t)256 * 9 * TRUNK_C * 2); // w1a|w1b
    unsigned short* w2r    = (unsigned short*)alloc((size_t)TRUNK_C * 9 * REG_C * 2);
    float*          regbuf = (float*)alloc(regN * 4);
    float*          gpconv = (float*)alloc(gpcN * 4);
    float*          gpool  = (float*)alloc((size_t)BATCH * 3 * GPOOL_C * 4);
    float*          gpbias = (float*)alloc((size_t)BATCH * REG_C * 4);
    unsigned short* act2   = (unsigned short*)alloc(regN * 2);                 // NHWC bf16

    // 1) trunk pre-activation -> NHWC bf16
    k_bn_mish_nhwc<<<(unsigned)((actN + 255) / 256), 256, 0, stream>>>(
        x, bn1_g, bn1_b, bn1_m, bn1_v, act1, actN);

    // 2) weights -> [OC][9][IC] bf16 ; w1a and w1b fused into one 256-row matrix
    int n1 = REG_C * TRUNK_C * 9;
    int n2 = GPOOL_C * TRUNK_C * 9;
    int n3 = TRUNK_C * REG_C * 9;
    k_wreorder<<<(n1 + 255) / 256, 256, 0, stream>>>(w1a, wfused, REG_C,   TRUNK_C, 0);
    k_wreorder<<<(n2 + 255) / 256, 256, 0, stream>>>(w1b, wfused, GPOOL_C, TRUNK_C, REG_C);
    k_wreorder<<<(n3 + 255) / 256, 256, 0, stream>>>(w2,  w2r,    TRUNK_C, REG_C,   0);

    // 3) fused conv1a+conv1b: OC=256 over shared im2col; rows 0-191 -> regbuf,
    //    rows 192-255 -> gpconv
    k_conv3x3_wmma<<<dim3(12, 256 / MT, BATCH), 256, 0, stream>>>(
        act1, wfused, regbuf, gpconv, nullptr, TRUNK_C, 256, REG_C);

    // 4) gpool: bn1b+mish fused into pooling
    k_gp_pool<<<BATCH * GPOOL_C, 128, 0, stream>>>(
        gpconv, b1b_g, b1b_b, b1b_m, b1b_v, gpool);

    // 5) linear bias
    k_gp_linear<<<(BATCH * REG_C + 255) / 256, 256, 0, stream>>>(
        gpool, w_lin, gpbias);

    // 6) reg + bias -> bn2 -> mish -> NHWC bf16
    k_bias_bn_mish_nhwc<<<(unsigned)((regN + 255) / 256), 256, 0, stream>>>(
        regbuf, gpbias, bn2_g, bn2_b, bn2_m, bn2_v, act2, regN);

    // 7) conv2 with fused residual add, straight to d_out (NCHW f32)
    k_conv3x3_wmma<<<dim3(12, TRUNK_C / MT, BATCH), 256, 0, stream>>>(
        act2, w2r, (float*)d_out, nullptr, x, REG_C, TRUNK_C, TRUNK_C);
}
